// MaxPoolingAggregator_55078660604188
// MI455X (gfx1250) — matrix operand order repaired and checked by reading, not compiled, not run on hardware
//
#include <hip/hip_runtime.h>

// MaxPoolingAggregator, fused single-pass kernel for MI455X (gfx1250).
// Memory-bound (1.76 GB traffic, ~29 GFLOP): read neigh_vecs exactly once,
// keep h in LDS, do all matmuls with v_wmma_f32_16x16x32_f16.

typedef __attribute__((ext_vector_type(16))) _Float16 v16h;
typedef __attribute__((ext_vector_type(8)))  float    v8f;

#define ALPHA        0.02f
#define NNODES       131072
#define NEIGHB       25
#define D_IN         128
#define D_HID        32
#define D_OUT        32
#define NODES_PER_WG 16
#define ROWS_PER_WG  (NODES_PER_WG * NEIGHB)  // 400
#define ROW_TILES    (ROWS_PER_WG / 16)       // 25
#define WAVES        8
#define LDH          (D_HID + 2)              // 34: pad so M=0..7 / M=8..15 halves hit disjoint LDS banks

__device__ __forceinline__ float leaky(float x) { return x >= 0.0f ? x : ALPHA * x; }

// A fragment (16x32 f16, MxK) for v_wmma_f32_16x16x32_f16 from row-major f32.
// ISA 7.12.2: lane L<16 holds row M=L, K=[0..8)+[16..24); lane L+16 holds
// row M=L, K=[8..16)+[24..32)  (within 32-wide K step `kk`).
__device__ __forceinline__ v16h load_a_f32(const float* base, int ldm, int kk, int lane) {
  const int m    = lane & 15;
  const int koff = kk * 32 + ((lane >> 4) << 3);
  const float* p = base + m * ldm + koff;
  v16h a;
#pragma unroll
  for (int i = 0; i < 8; ++i) a[i] = (_Float16)p[i];
#pragma unroll
  for (int i = 0; i < 8; ++i) a[8 + i] = (_Float16)p[16 + i];
  return a;
}

// B fragment (32x16 f16, KxN) from row-major f32 W[K][ldn]: lane L holds
// column N=nbase+(L&15); lanes 0-15 carry K=[0..8)+[16..24), lanes 16-31
// K=[8..16)+[24..32) (same K order as the A fragment).
__device__ __forceinline__ v16h load_b_f32(const float* W, int ldn, int kk, int nbase, int lane) {
  const int n  = nbase + (lane & 15);
  const int k0 = kk * 32 + ((lane >> 4) << 3);
  v16h b;
#pragma unroll
  for (int i = 0; i < 8; ++i) b[i] = (_Float16)W[(k0 + i) * ldn + n];
#pragma unroll
  for (int i = 0; i < 8; ++i) b[8 + i] = (_Float16)W[(k0 + 16 + i) * ldn + n];
  return b;
}

__global__ __launch_bounds__(256) void MaxPoolingAggregator_55078660604188_kernel(
    const float* __restrict__ self_vecs,   // [N, 128]
    const float* __restrict__ neigh_vecs,  // [N, 25, 128]
    const float* __restrict__ W_mlp,       // [128, 32]
    const float* __restrict__ b_mlp,       // [32]
    const float* __restrict__ W_va,        // [128, 32]
    const float* __restrict__ W_neigh,     // [32, 32]
    float* __restrict__ out) {             // [N, 32]
  __shared__ float sh_h[ROWS_PER_WG][LDH];          // 400 x 34 f32 = 54.4 KB
  __shared__ float sh_pool[NODES_PER_WG][D_HID];    // 16 x 32 f32

  const int lane  = threadIdx.x & 31;
  const int wave  = threadIdx.x >> 5;
  const int node0 = blockIdx.x * NODES_PER_WG;
  const float* Ablk = neigh_vecs + (size_t)node0 * NEIGHB * D_IN;

  // ---------- Phase 1: h = leaky(neigh @ W_mlp + b), staged into LDS ----------
  // Cache all W_mlp B-fragments in VGPRs (2 col tiles x 4 K-steps).
  v16h bm[2][4];
#pragma unroll
  for (int ct = 0; ct < 2; ++ct)
#pragma unroll
    for (int kk = 0; kk < 4; ++kk)
      bm[ct][kk] = load_b_f32(W_mlp, D_HID, kk, ct * 16, lane);

  const int nn    = lane & 15;          // C/D column within tile
  const int m_off = (lane >> 4) << 3;   // C/D row offset (0 or 8)
  const float bias0 = b_mlp[nn];
  const float bias1 = b_mlp[16 + nn];

  for (int rt = wave; rt < ROW_TILES; rt += WAVES) {
    const float* arow = Ablk + rt * 16 * D_IN;
    if (rt + WAVES < ROW_TILES)   // gfx1250 global_prefetch_b8: next tile (8 KB / 32 lanes = 256B each)
      __builtin_prefetch(arow + WAVES * 16 * D_IN + lane * 64, 0, 1);

    v16h a[4];
#pragma unroll
    for (int kk = 0; kk < 4; ++kk) a[kk] = load_a_f32(arow, D_IN, kk, lane);

    v8f c0 = {};
    v8f c1 = {};
#pragma unroll
    for (int kk = 0; kk < 4; ++kk) {
      c0 = __builtin_amdgcn_wmma_f32_16x16x32_f16(false, a[kk], false, bm[0][kk],
                                                  (short)0, c0, false, false);
      c1 = __builtin_amdgcn_wmma_f32_16x16x32_f16(false, a[kk], false, bm[1][kk],
                                                  (short)0, c1, false, false);
    }
#pragma unroll
    for (int v = 0; v < 8; ++v) {
      const int r = rt * 16 + v + m_off;
      sh_h[r][nn]      = leaky(c0[v] + bias0);
      sh_h[r][16 + nn] = leaky(c1[v] + bias1);
    }
  }
  __syncthreads();

  // ---------- Phase 2: max-pool over the 25 neighbors of each node ----------
  for (int o = threadIdx.x; o < NODES_PER_WG * D_HID; o += blockDim.x) {
    const int node = o >> 5;
    const int c    = o & 31;
    float m = -3.402823466e38f;
#pragma unroll
    for (int j = 0; j < NEIGHB; ++j) m = fmaxf(m, sh_h[node * NEIGHB + j][c]);
    sh_pool[node][c] = m;
  }
  __syncthreads();

  // ---------- Phase 3: out = leaky(self@W_va + pooled@W_neigh) ----------
  // One 16x32 output block per WG; waves 0 and 1 own one 16-col tile each.
  if (wave < 2) {
    const int ct = wave;
    v8f acc = {};
    const float* S = self_vecs + (size_t)node0 * D_IN;
#pragma unroll
    for (int kk = 0; kk < 4; ++kk) {
      v16h a = load_a_f32(S, D_IN, kk, lane);
      v16h b = load_b_f32(W_va, D_OUT, kk, ct * 16, lane);
      acc = __builtin_amdgcn_wmma_f32_16x16x32_f16(false, a, false, b,
                                                   (short)0, acc, false, false);
    }
    {  // pooled (16x32) @ W_neigh (32x32): single K=32 step, same accumulator => fused add
      v16h a = load_a_f32(&sh_pool[0][0], D_HID, 0, lane);
      v16h b = load_b_f32(W_neigh, D_OUT, 0, ct * 16, lane);
      acc = __builtin_amdgcn_wmma_f32_16x16x32_f16(false, a, false, b,
                                                   (short)0, acc, false, false);
    }
#pragma unroll
    for (int v = 0; v < 8; ++v) {
      const int row = node0 + v + m_off;
      out[row * D_OUT + ct * 16 + nn] = leaky(acc[v]);
    }
  }
}

extern "C" void kernel_launch(void* const* d_in, const int* in_sizes, int n_in,
                              void* d_out, int out_size, void* d_ws, size_t ws_size,
                              hipStream_t stream) {
  const float* self_vecs  = (const float*)d_in[0];
  const float* neigh_vecs = (const float*)d_in[1];
  const float* W_mlp      = (const float*)d_in[2];
  const float* b_mlp      = (const float*)d_in[3];
  const float* W_va       = (const float*)d_in[4];
  const float* W_neigh    = (const float*)d_in[5];
  float* out = (float*)d_out;

  dim3 grid(NNODES / NODES_PER_WG);  // 8192 WGs
  dim3 block(256);                   // 8 wave32 waves
  hipLaunchKernelGGL(MaxPoolingAggregator_55078660604188_kernel, grid, block, 0, stream,
                     self_vecs, neigh_vecs, W_mlp, b_mlp, W_va, W_neigh, out);
}
